// ScaledDotProductAttention_841813590085
// MI455X (gfx1250) — matrix operand order repaired and checked by reading, not compiled, not run on hardware
//
#include <hip/hip_runtime.h>

#define TQ_ 2048
#define TK_ 2048
#define DH_ 64
#define NB_ 64
#define BQ_ 16

typedef _Float16 v16h __attribute__((ext_vector_type(16)));
typedef _Float16 h8   __attribute__((ext_vector_type(8)));
typedef float    v8f  __attribute__((ext_vector_type(8)));
typedef float    f4v  __attribute__((ext_vector_type(4)));

__launch_bounds__(256, 1)
__global__ void sdpa_wmma_kernel(const float* __restrict__ Q,
                                 const float* __restrict__ K,
                                 const float* __restrict__ V,
                                 const unsigned char* __restrict__ Mk,
                                 float* __restrict__ outO,
                                 float* __restrict__ outA)
{
    // Big-LDS design: full 16x2048 f32 score tile lives in LDS (CDNA5: 320KB/WGP).
    __shared__ float    sS[16][2052];                 // 131,328 B (padded stride)
    __shared__ _Float16 sQ[16][64];                   //   2,048 B
    __shared__ __align__(16) unsigned char sKV[33792];// K stage 128x72 h / Vt stage 64x264 h
    _Float16 (*sK)[72]   = (_Float16 (*)[72])sKV;
    _Float16 (*sVt)[264] = (_Float16 (*)[264])sKV;

    const int t    = threadIdx.x;
    const int lane = t & 31;
    const int w    = t >> 5;          // wave id 0..7
    const int hi   = (lane >> 4) & 1; // lane half
    const int ln   = lane & 15;
    const int b    = blockIdx.y;
    const int q0   = blockIdx.x * BQ_;

    const float* Qb = Q + ((size_t)b * TQ_ + q0) * DH_;
    const float* Kb = K + (size_t)b * TK_ * DH_;
    const float* Vb = V + (size_t)b * TK_ * DH_;

    // ---- stage Q tile (16x64 f32 -> f16), contiguous 1024 floats ----
    {
        float4 v = ((const float4*)Qb)[t];
        int row = (t * 4) >> 6, d = (t * 4) & 63;
        sQ[row][d + 0] = (_Float16)v.x; sQ[row][d + 1] = (_Float16)v.y;
        sQ[row][d + 2] = (_Float16)v.z; sQ[row][d + 3] = (_Float16)v.w;
    }
    __syncthreads();

    // ---- build A(Q) operands for the two 32-wide k windows (held in VGPRs) ----
    v16h aQ[2];
#pragma unroll
    for (int kw = 0; kw < 2; ++kw) {
        const int g0 = kw * 32 + (hi ? 16 : 0);
        h8 lo = *(const h8*)&sQ[ln][g0];
        h8 hh = *(const h8*)&sQ[ln][g0 + 8];
        v16h a;
#pragma unroll
        for (int i = 0; i < 8; ++i) { a[i] = lo[i]; a[i + 8] = hh[i]; }
        aQ[kw] = a;
    }

    // ================= Phase 1: S = Q*K^T * (1/8), tiles into LDS =================
    for (int kb = 0; kb < TK_; kb += 128) {
        __syncthreads();
        if (kb + 128 < TK_) {   // prefetch next K chunk (global_prefetch_b8)
            __builtin_prefetch(Kb + (size_t)(kb + 128) * DH_ + t * 16, 0, 0);
            __builtin_prefetch(Kb + (size_t)(kb + 128) * DH_ + 4096 + t * 16, 0, 0);
        }
        // stage K chunk [128 keys][64 dims] f32 -> f16 row-major
        const float4* ks = (const float4*)(Kb + (size_t)kb * DH_);
#pragma unroll
        for (int i = 0; i < 8; ++i) {
            int idx = t * 4 + i * 1024;
            float4 v = ks[idx >> 2];
            int key = idx >> 6, d = idx & 63;
            sK[key][d + 0] = (_Float16)v.x; sK[key][d + 1] = (_Float16)v.y;
            sK[key][d + 2] = (_Float16)v.z; sK[key][d + 3] = (_Float16)v.w;
        }
        __syncthreads();

        // each wave computes a 16(q) x 16(key) f32 tile: two f16 WMMAs over k=64.
        // Issue all four B-operand LDS loads before the first WMMA so the second
        // WMMA's operands are already in flight (no mid-tile s_wait_dscnt 0 stall).
        const int key = w * 16 + ln;        // B-operand: n = lane%16
        const int kh  = hi * 16;            // lanes 16-31 carry k=16..31 of window
        h8 b00 = *(const h8*)&sK[key][0  + kh];
        h8 b01 = *(const h8*)&sK[key][0  + kh + 8];
        h8 b10 = *(const h8*)&sK[key][32 + kh];
        h8 b11 = *(const h8*)&sK[key][32 + kh + 8];
        v16h bb0, bb1;
#pragma unroll
        for (int i = 0; i < 8; ++i) {
            bb0[i] = b00[i]; bb0[i + 8] = b01[i];
            bb1[i] = b10[i]; bb1[i + 8] = b11[i];
        }
        v8f c = {};
        c = __builtin_amdgcn_wmma_f32_16x16x32_f16(false, aQ[0], false, bb0,
                                                   (short)0, c, false, false);
        c = __builtin_amdgcn_wmma_f32_16x16x32_f16(false, aQ[1], false, bb1,
                                                   (short)0, c, false, false);
        const int col = kb + w * 16 + ln;
#pragma unroll
        for (int r = 0; r < 8; ++r)
            sS[r + hi * 8][col] = c[r] * 0.125f;   // 1/sqrt(64)
    }
    __syncthreads();

    // ================= Phase 2: mask + softmax over key axis, write attn =========
    const unsigned char* Mb = Mk + ((size_t)b * TQ_ + q0) * TK_;
    float* Ab = outA + ((size_t)b * TQ_ + q0) * TK_;
#pragma unroll
    for (int rr = 0; rr < 2; ++rr) {
        const int m = w * 2 + rr;
        const unsigned char* mrow = Mb + (size_t)m * TK_;
        float mx = -3.0e38f;
        for (int it = 0; it < 16; ++it) {
            const int cx = it * 128 + lane * 4;
            float4 s = *(float4*)&sS[m][cx];
            // mask bytes are single-touch: stream them past L2 (non-temporal)
            unsigned int mk = __builtin_nontemporal_load((const unsigned int*)(mrow + cx));
            s.x = (mk & 0x000000ffu) ? -1.0e18f : s.x;
            s.y = (mk & 0x0000ff00u) ? -1.0e18f : s.y;
            s.z = (mk & 0x00ff0000u) ? -1.0e18f : s.z;
            s.w = (mk & 0xff000000u) ? -1.0e18f : s.w;
            *(float4*)&sS[m][cx] = s;
            mx = fmaxf(mx, fmaxf(fmaxf(s.x, s.y), fmaxf(s.z, s.w)));
        }
#pragma unroll
        for (int o = 16; o >= 1; o >>= 1) mx = fmaxf(mx, __shfl_xor(mx, o, 32));
        float sum = 0.0f;
        for (int it = 0; it < 16; ++it) {
            const int cx = it * 128 + lane * 4;
            float4 s = *(float4*)&sS[m][cx];
            s.x = __expf(s.x - mx); s.y = __expf(s.y - mx);
            s.z = __expf(s.z - mx); s.w = __expf(s.w - mx);
            *(float4*)&sS[m][cx] = s;
            sum += (s.x + s.y) + (s.z + s.w);
        }
#pragma unroll
        for (int o = 16; o >= 1; o >>= 1) sum += __shfl_xor(sum, o, 32);
        const float inv = __frcp_rn(sum);
        float* arow = Ab + (size_t)m * TK_;
        for (int it = 0; it < 16; ++it) {
            const int cx = it * 128 + lane * 4;
            float4 s = *(float4*)&sS[m][cx];
            s.x *= inv; s.y *= inv; s.z *= inv; s.w *= inv;
            *(float4*)&sS[m][cx] = s;            // P kept in LDS for PV
            // attn is written once and never re-read: non-temporal b128 store,
            // keeps the 192MB L2 free for the reused K/V operands.
            f4v pv = { s.x, s.y, s.z, s.w };
            __builtin_nontemporal_store(pv, (f4v*)(arow + cx));
        }
    }
    __syncthreads();

    // ================= Phase 3: O = P * V (wave split: 4 dim-tiles x 2 key halves)
    v8f oAcc = {};
    const int nTile = w & 3;
    const int kh2   = (w >> 2) * 128;
    for (int cb = 0; cb < TK_; cb += 256) {
        __syncthreads();
        if (cb + 256 < TK_) {   // prefetch next V chunk
            __builtin_prefetch(Vb + (size_t)(cb + 256) * DH_ + t * 16, 0, 0);
            __builtin_prefetch(Vb + (size_t)(cb + 256) * DH_ +  4096 + t * 16, 0, 0);
            __builtin_prefetch(Vb + (size_t)(cb + 256) * DH_ +  8192 + t * 16, 0, 0);
            __builtin_prefetch(Vb + (size_t)(cb + 256) * DH_ + 12288 + t * 16, 0, 0);
        }
        // stage V chunk [256 keys][64 dims] f32, transposed -> sVt[64][264] f16
        const float4* vs = (const float4*)(Vb + (size_t)cb * DH_);
#pragma unroll
        for (int i = 0; i < 16; ++i) {
            int idx = t * 4 + i * 1024;
            float4 v = vs[idx >> 2];
            int key = idx >> 6, d = idx & 63;
            sVt[d + 0][key] = (_Float16)v.x; sVt[d + 1][key] = (_Float16)v.y;
            sVt[d + 2][key] = (_Float16)v.z; sVt[d + 3][key] = (_Float16)v.w;
        }
        __syncthreads();
#pragma unroll
        for (int kk = 0; kk < 128; kk += 32) {
            const int kAbs = cb + kh2 + kk;
            // A operand: P rows from LDS (f32 -> f16 on the fly)
            const int g0 = kAbs + (hi ? 8 : 0);
            float4 x0 = *(const float4*)&sS[ln][g0];
            float4 x1 = *(const float4*)&sS[ln][g0 + 4];
            float4 y0 = *(const float4*)&sS[ln][g0 + 16];
            float4 y1 = *(const float4*)&sS[ln][g0 + 20];
            v16h a;
            a[0]  = (_Float16)x0.x; a[1]  = (_Float16)x0.y; a[2]  = (_Float16)x0.z; a[3]  = (_Float16)x0.w;
            a[4]  = (_Float16)x1.x; a[5]  = (_Float16)x1.y; a[6]  = (_Float16)x1.z; a[7]  = (_Float16)x1.w;
            a[8]  = (_Float16)y0.x; a[9]  = (_Float16)y0.y; a[10] = (_Float16)y0.z; a[11] = (_Float16)y0.w;
            a[12] = (_Float16)y1.x; a[13] = (_Float16)y1.y; a[14] = (_Float16)y1.z; a[15] = (_Float16)y1.w;
            // B operand: transposed V columns, contiguous in sVt
            const int dim    = nTile * 16 + ln;
            const int klocal = kh2 + kk + hi * 8;
            h8 b0 = *(const h8*)&sVt[dim][klocal];
            h8 b1 = *(const h8*)&sVt[dim][klocal + 16];
            v16h bb;
#pragma unroll
            for (int i = 0; i < 8; ++i) { bb[i] = b0[i]; bb[i + 8] = b1[i]; }
            oAcc = __builtin_amdgcn_wmma_f32_16x16x32_f16(false, a, false, bb,
                                                          (short)0, oAcc, false, false);
        }
    }
    __syncthreads();
    // reduce the two key-halves through LDS (reuse sS), then store O
    float* sO = &sS[0][0];
    if (w >= 4) {
#pragma unroll
        for (int r = 0; r < 8; ++r) sO[(w - 4) * 256 + r * 32 + lane] = oAcc[r];
    }
    __syncthreads();
    if (w < 4) {
        float* Ob = outO + ((size_t)b * TQ_ + q0) * DH_;
#pragma unroll
        for (int r = 0; r < 8; ++r) {
            float vsum = oAcc[r] + sO[w * 256 + r * 32 + lane];
            __builtin_nontemporal_store(
                vsum, Ob + (size_t)(r + hi * 8) * DH_ + nTile * 16 + ln);
        }
    }
}

extern "C" void kernel_launch(void* const* d_in, const int* in_sizes, int n_in,
                              void* d_out, int out_size, void* d_ws, size_t ws_size,
                              hipStream_t stream) {
    (void)in_sizes; (void)n_in; (void)out_size; (void)d_ws; (void)ws_size;
    const float* q = (const float*)d_in[0];
    const float* k = (const float*)d_in[1];
    const float* v = (const float*)d_in[2];
    const unsigned char* m = (const unsigned char*)d_in[3]; // jnp bool = 1 byte
    float* outO = (float*)d_out;                             // [B,TQ,D] first
    float* outA = outO + (size_t)NB_ * TQ_ * DH_;            // then [B,TQ,TK]
    dim3 grid(TQ_ / BQ_, NB_);
    sdpa_wmma_kernel<<<grid, 256, 0, stream>>>(q, k, v, m, outO, outA);
}
